// MixtralSparseMoeBlock_66718021976274
// MI455X (gfx1250) — compile-verified, matrix operand order checked by reading, not run on hardware
//
#include <hip/hip_runtime.h>
#include <hip/hip_bf16.h>
#include <stdint.h>

// Problem constants (from reference): B=1, S=2048, D=1024, E=8, K=2, FF=4096
#define TT 2048
#define DD 1024
#define EE 8
#define KTOP 2
#define FFD 4096
#define MAX_ROWS 4352  // T*K + E*31 padding headroom, rounded up

typedef __attribute__((ext_vector_type(8)))  float  v8f;
typedef __attribute__((ext_vector_type(16))) __bf16 v16bf;

// ---------------------------------------------------------------------------
// WMMA helpers: D = A(16x32 bf16) x B(32x16 bf16) + C(16x16 f32)
// A-fragment per ISA 7.12.2: lane L holds row r=L&15; K-halves split by L>>4:
//   VGPR0..3 = K in [half*8, half*8+8), VGPR4..7 = K in [16+half*8, 16+half*8+8)
// B-fragment mirrors with N<->M (column n = lane&15), so a row-major [N, D]
// weight matrix row loads with the exact same per-lane pattern as A.
// ---------------------------------------------------------------------------
__device__ __forceinline__ v16bf load_frag_bf16(const __bf16* p) {
    union { v16bf v; uint4 q[2]; } u;
    u.q[0] = *(const uint4*)(p);        // 8 bf16 : K = base .. base+7
    u.q[1] = *(const uint4*)(p + 16);   // 8 bf16 : K = base+16 .. base+23
    return u.v;
}

__device__ __forceinline__ v16bf load_frag_f32_cvt(const float* p) {
    float4 f0 = *(const float4*)(p);
    float4 f1 = *(const float4*)(p + 4);
    float4 f2 = *(const float4*)(p + 16);
    float4 f3 = *(const float4*)(p + 20);
    v16bf v;
    v[0]  = (__bf16)f0.x; v[1]  = (__bf16)f0.y; v[2]  = (__bf16)f0.z; v[3]  = (__bf16)f0.w;
    v[4]  = (__bf16)f1.x; v[5]  = (__bf16)f1.y; v[6]  = (__bf16)f1.z; v[7]  = (__bf16)f1.w;
    v[8]  = (__bf16)f2.x; v[9]  = (__bf16)f2.y; v[10] = (__bf16)f2.z; v[11] = (__bf16)f2.w;
    v[12] = (__bf16)f3.x; v[13] = (__bf16)f3.y; v[14] = (__bf16)f3.z; v[15] = (__bf16)f3.w;
    return v;
}

__device__ __forceinline__ v8f wmma_bf16(v16bf a, v16bf b, v8f c) {
    return __builtin_amdgcn_wmma_f32_16x16x32_bf16(
        /*neg_a=*/false, a, /*neg_b=*/false, b,
        /*c_mod=*/(short)0, c, /*reuse_a=*/false, /*reuse_b=*/false);
}

// silu(u)*g with hardware reciprocal (avoid the IEEE div expansion)
__device__ __forceinline__ float silu_mul(float u, float g) {
    return u * __builtin_amdgcn_rcpf(1.0f + __expf(-u)) * g;
}

// ---------------------------------------------------------------------------
// Kernel 1: fp32 -> bf16 activation conversion (T*D elements, x4 vectorized)
// ---------------------------------------------------------------------------
__global__ void moe_cvt_bf16(const float* __restrict__ h, __bf16* __restrict__ hbf) {
    size_t i = ((size_t)blockIdx.x * blockDim.x + threadIdx.x) * 4;
    float4 f = *(const float4*)(h + i);
    union { __bf16 b[4]; unsigned long long u; } u;
    u.b[0] = (__bf16)f.x; u.b[1] = (__bf16)f.y; u.b[2] = (__bf16)f.z; u.b[3] = (__bf16)f.w;
    *(unsigned long long*)(hbf + i) = u.u;
}

// ---------------------------------------------------------------------------
// Kernel 2: router — one wave32 per token. logits = h @ gate_w^T, softmax,
// top-2, renormalize. Writes raw logits to output tail + (idx, w) to ws.
// ---------------------------------------------------------------------------
__global__ void moe_router(const float* __restrict__ h, const float* __restrict__ gw,
                           float* __restrict__ logits_out,
                           int* __restrict__ top_idx, float* __restrict__ top_w) {
    const int wid  = threadIdx.x >> 5;
    const int lane = threadIdx.x & 31;
    const int t    = blockIdx.x * 8 + wid;

    float acc[EE];
#pragma unroll
    for (int e = 0; e < EE; ++e) acc[e] = 0.0f;

    for (int i = 0; i < DD / 32; ++i) {
        const int k = i * 32 + lane;
        const float hv = h[(size_t)t * DD + k];
#pragma unroll
        for (int e = 0; e < EE; ++e) acc[e] += hv * gw[e * DD + k];
    }
#pragma unroll
    for (int e = 0; e < EE; ++e) {
        for (int off = 16; off > 0; off >>= 1) acc[e] += __shfl_xor(acc[e], off);
    }
    if (lane == 0) {
        float m = acc[0];
#pragma unroll
        for (int e = 1; e < EE; ++e) m = fmaxf(m, acc[e]);
        float rw[EE]; float s = 0.0f;
#pragma unroll
        for (int e = 0; e < EE; ++e) { rw[e] = __expf(acc[e] - m); s += rw[e]; }
        const float inv = 1.0f / s;
#pragma unroll
        for (int e = 0; e < EE; ++e) {
            rw[e] *= inv;
            logits_out[(size_t)t * EE + e] = acc[e];  // raw logits (second output)
        }
        int i0 = 0;
#pragma unroll
        for (int e = 1; e < EE; ++e) if (rw[e] > rw[i0]) i0 = e;
        int i1 = (i0 == 0) ? 1 : 0;
#pragma unroll
        for (int e = 0; e < EE; ++e) if (e != i0 && rw[e] > rw[i1]) i1 = e;
        const float ws2 = rw[i0] + rw[i1];
        top_idx[t * 2 + 0] = i0;  top_w[t * 2 + 0] = rw[i0] / ws2;
        top_idx[t * 2 + 1] = i1;  top_w[t * 2 + 1] = rw[i1] / ws2;
    }
}

// ---------------------------------------------------------------------------
// Kernel 3: deterministic gather — one wave per expert, ballot prefix-scan in
// token order. Builds compact per-expert (token, weight, slot) lists + bases.
// ---------------------------------------------------------------------------
__global__ void moe_gather(const int* __restrict__ top_idx, const float* __restrict__ top_w,
                           int* __restrict__ counts, int* __restrict__ base,
                           int* __restrict__ tok_list, float* __restrict__ w_list,
                           int* __restrict__ slot_list) {
    const int e    = threadIdx.x >> 5;
    const int lane = threadIdx.x & 31;
    int cnt = 0;
    for (int c = 0; c < TT; c += 32) {
        const int t  = c + lane;
        const int i0 = top_idx[t * 2 + 0];
        const int i1 = top_idx[t * 2 + 1];
        const int slot = (i0 == e) ? 0 : ((i1 == e) ? 1 : -1);
        const unsigned long long mask = __ballot(slot >= 0);
        if (slot >= 0) {
            const int p = cnt + __popcll(mask & ((1ull << lane) - 1ull));
            tok_list[e * TT + p]  = t;
            w_list[e * TT + p]    = top_w[t * 2 + slot];
            slot_list[e * TT + p] = slot;
        }
        cnt += __popcll(mask);
    }
    if (lane == 0) counts[e] = cnt;
    __syncthreads();
    if (threadIdx.x == 0) {
        int b = 0;
        for (int ee = 0; ee < EE; ++ee) {
            base[ee] = b;
            b += ((counts[ee] + 31) >> 5) << 5;  // pad rows to 32
        }
        base[EE] = b;
    }
}

// ---------------------------------------------------------------------------
// Kernel 4: grouped up/gate GEMM + SiLU. Block = 128 thr (4 waves).
// Tile: M=64 gathered tokens (4 A-fragments), N=64 FF cols (16 per wave),
// K=D=1024. Per K-step: 2 converted B-fragments amortized over 4 A-fragments
// -> 8 independent v_wmma per iteration (hides WMMA->WMMA RAW latency).
// inter[base[e]+row, ff] = silu(x@w1^T) * (x@w3^T)  in bf16.
// Weight reuse across row-tiles is served by the 192MB L2.
// ---------------------------------------------------------------------------
__global__ void moe_up_gate(const float* __restrict__ w1, const float* __restrict__ w3,
                            const __bf16* __restrict__ hbf,
                            const int* __restrict__ tok_list, const int* __restrict__ counts,
                            const int* __restrict__ base, __bf16* __restrict__ inter) {
    const int e = blockIdx.y;
    const int cnt = counts[e];
    const int rowTile = blockIdx.z * 64;
    if (rowTile >= cnt) return;

    const int wave = threadIdx.x >> 5;
    const int lane = threadIdx.x & 31;
    const int r    = lane & 15;
    const int half = lane >> 4;

    const __bf16* ap[4];
#pragma unroll
    for (int i = 0; i < 4; ++i) {
        const int t = tok_list[e * TT + min(rowTile + i * 16 + r, cnt - 1)];
        ap[i] = hbf + (size_t)t * DD;
    }

    const int f = blockIdx.x * 64 + wave * 16 + r;          // W1/W3 row = B column
    const float* w1p = w1 + ((size_t)e * FFD + f) * DD;
    const float* w3p = w3 + ((size_t)e * FFD + f) * DD;

    v8f accU[4] = {{}, {}, {}, {}};
    v8f accG[4] = {{}, {}, {}, {}};
    for (int k0 = 0; k0 < DD; k0 += 32) {
        const int kA = k0 + half * 8;
        __builtin_prefetch(w1p + k0 + 256, 0, 0);
        __builtin_prefetch(w3p + k0 + 256, 0, 0);
        const v16bf b1 = load_frag_f32_cvt(w1p + kA);
        const v16bf b3 = load_frag_f32_cvt(w3p + kA);
#pragma unroll
        for (int i = 0; i < 4; ++i) {
            const v16bf a = load_frag_bf16(ap[i] + kA);
            accU[i] = wmma_bf16(a, b1, accU[i]);
            accG[i] = wmma_bf16(a, b3, accG[i]);
        }
    }

    const int eb = base[e];
    const int ncol = blockIdx.x * 64 + wave * 16 + r;       // N = lane&15
#pragma unroll
    for (int i = 0; i < 4; ++i) {
#pragma unroll
        for (int j = 0; j < 8; ++j) {                       // M = j + 8*half
            const int m = rowTile + i * 16 + j + 8 * half;
            if (m < cnt)
                inter[(size_t)(eb + m) * FFD + ncol] = (__bf16)silu_mul(accU[i][j], accG[i][j]);
        }
    }
}

// ---------------------------------------------------------------------------
// Kernel 5: grouped down GEMM. Tile: M=64 rows (4 A-fragments), N=64 D cols,
// K=FF=4096. One converted W2 B-fragment feeds 4 independent v_wmma per step.
// Scales by routing weight, scatters fp32 into per-(token,slot) buffer:
// every (t,slot) written exactly once -> deterministic, no atomics.
// ---------------------------------------------------------------------------
__global__ void moe_down(const float* __restrict__ w2, const __bf16* __restrict__ inter,
                         const int* __restrict__ tok_list, const float* __restrict__ w_list,
                         const int* __restrict__ slot_list, const int* __restrict__ counts,
                         const int* __restrict__ base, float* __restrict__ outc) {
    const int e = blockIdx.y;
    const int cnt = counts[e];
    const int rowTile = blockIdx.z * 64;
    if (rowTile >= cnt) return;

    const int wave = threadIdx.x >> 5;
    const int lane = threadIdx.x & 31;
    const int r    = lane & 15;
    const int half = lane >> 4;

    const int eb = base[e];
    const __bf16* ap[4];
#pragma unroll
    for (int i = 0; i < 4; ++i) {
        const int ar = eb + min(rowTile + i * 16 + r, cnt - 1);
        ap[i] = inter + (size_t)ar * FFD;
    }

    const int d = blockIdx.x * 64 + wave * 16 + r;          // W2 row = B column
    const float* w2p = w2 + ((size_t)e * DD + d) * FFD;

    v8f acc[4] = {{}, {}, {}, {}};
    for (int k0 = 0; k0 < FFD; k0 += 32) {
        const int kA = k0 + half * 8;
        __builtin_prefetch(w2p + k0 + 256, 0, 0);
        const v16bf b = load_frag_f32_cvt(w2p + kA);
#pragma unroll
        for (int i = 0; i < 4; ++i) {
            const v16bf a = load_frag_bf16(ap[i] + kA);
            acc[i] = wmma_bf16(a, b, acc[i]);
        }
    }

    const int ncol = blockIdx.x * 64 + wave * 16 + r;
#pragma unroll
    for (int i = 0; i < 4; ++i) {
#pragma unroll
        for (int j = 0; j < 8; ++j) {
            const int m = rowTile + i * 16 + j + 8 * half;
            if (m < cnt) {
                const int p = e * TT + m;
                const int t = tok_list[p];
                outc[((size_t)t * KTOP + slot_list[p]) * DD + ncol] = w_list[p] * acc[i][j];
            }
        }
    }
}

// ---------------------------------------------------------------------------
// Kernel 6: combine the two expert-slot contributions per token.
// ---------------------------------------------------------------------------
__global__ void moe_combine(const float* __restrict__ outc, float* __restrict__ out) {
    size_t i = ((size_t)blockIdx.x * blockDim.x + threadIdx.x) * 4;
    const size_t t = i / DD;
    const size_t d = i % DD;
    const float4 a = *(const float4*)(outc + (t * KTOP + 0) * DD + d);
    const float4 b = *(const float4*)(outc + (t * KTOP + 1) * DD + d);
    float4 rv;
    rv.x = a.x + b.x; rv.y = a.y + b.y; rv.z = a.z + b.z; rv.w = a.w + b.w;
    *(float4*)(out + i) = rv;
}

// ---------------------------------------------------------------------------
extern "C" void kernel_launch(void* const* d_in, const int* in_sizes, int n_in,
                              void* d_out, int out_size, void* d_ws, size_t ws_size,
                              hipStream_t stream) {
    const float* h   = (const float*)d_in[0];   // [T, D]
    const float* gw  = (const float*)d_in[1];   // [E, D]
    const float* w1  = (const float*)d_in[2];   // [E, FF, D]
    const float* w3  = (const float*)d_in[3];   // [E, FF, D]
    const float* w2  = (const float*)d_in[4];   // [E, D, FF]

    float* out    = (float*)d_out;              // [T, D]
    float* logits = out + (size_t)TT * DD;      // [T, E]

    // workspace carve (256B aligned)
    uintptr_t p = (uintptr_t)d_ws;
    auto carve = [&](size_t bytes) -> void* {
        void* r = (void*)p;
        p += (bytes + 255) & ~(size_t)255;
        return r;
    };
    __bf16* hbf      = (__bf16*)carve((size_t)TT * DD * sizeof(__bf16));
    int*    top_idx  = (int*)   carve((size_t)TT * KTOP * sizeof(int));
    float*  top_w    = (float*) carve((size_t)TT * KTOP * sizeof(float));
    int*    counts   = (int*)   carve(EE * sizeof(int));
    int*    base     = (int*)   carve((EE + 1) * sizeof(int));
    int*    tok_list = (int*)   carve((size_t)EE * TT * sizeof(int));
    float*  w_list   = (float*) carve((size_t)EE * TT * sizeof(float));
    int*    slot_lst = (int*)   carve((size_t)EE * TT * sizeof(int));
    __bf16* inter    = (__bf16*)carve((size_t)MAX_ROWS * FFD * sizeof(__bf16));
    float*  outc     = (float*) carve((size_t)TT * KTOP * DD * sizeof(float));

    // 1) activations -> bf16
    moe_cvt_bf16<<<(TT * DD) / (256 * 4), 256, 0, stream>>>(h, hbf);
    // 2) router (one wave per token)
    moe_router<<<TT / 8, 256, 0, stream>>>(h, gw, logits, top_idx, top_w);
    // 3) deterministic per-expert gather + base prefix
    moe_gather<<<1, EE * 32, 0, stream>>>(top_idx, top_w, counts, base,
                                          tok_list, w_list, slot_lst);
    // 4) up/gate grouped GEMM + SiLU  (grid: ff-tiles x experts x row-tiles)
    moe_up_gate<<<dim3(FFD / 64, EE, TT / 64), 128, 0, stream>>>(
        w1, w3, hbf, tok_list, counts, base, inter);
    // 5) down grouped GEMM, scaled scatter
    moe_down<<<dim3(DD / 64, EE, TT / 64), 128, 0, stream>>>(
        w2, inter, tok_list, w_list, slot_lst, counts, base, outc);
    // 6) combine top-2 contributions
    moe_combine<<<(TT * DD) / (256 * 4), 256, 0, stream>>>(outc, out);
}
